// SPDGMultiHeadAttention_48593259987327
// MI455X (gfx1250) — matrix operand order repaired and checked by reading, not compiled, not run on hardware
//
#include <hip/hip_runtime.h>
#include <hip/hip_bf16.h>

// ---------------------------------------------------------------------------
// CDNA5 (gfx1250) transformer block: LN -> QKV -> MHA -> O+res -> LN -> FFN+res
// Matmuls: v_wmma_f32_16x16x32_bf16 (fp32 accumulate).
// GEMM tile staging: Tensor Data Mover (tensor_load_to_lds) + TENSORcnt,
// double-buffered in LDS.
// ---------------------------------------------------------------------------

typedef __attribute__((ext_vector_type(16))) __bf16 v16bf;
typedef __attribute__((ext_vector_type(8)))  __bf16 v8bf;
typedef __attribute__((ext_vector_type(8)))  float  v8f;
typedef __attribute__((ext_vector_type(4)))  unsigned int u32x4;
typedef __attribute__((ext_vector_type(8)))  int i32x8;
typedef __attribute__((ext_vector_type(4)))  int i32x4;

#define LDA 40   // padded LDS row stride in bf16 elems (64B row + 16B pad = 80B = 5*16B)

__device__ __forceinline__ float gelu_exact(float x) {
    return 0.5f * x * (1.0f + erff(x * 0.70710678118654752f));
}

// ---------------------------------------------------------------------------
// TDM: issue a 2-D tensor_load_to_lds (6-arg clang-23 form).
//   - element size 2B (bf16), tile rows of tile_d0 elems, tile_d1 rows
//   - LDS padding: 16 DWORDs data (64B row) then 4 DWORDs pad -> LDA=40 elems
//   D# group0: count=1 | lds_addr | global_addr(57b) | type=2
//   D# group1: data_size=1(2B), pad_enable, pad_interval=3(16dw), pad_amount=3(4dw),
//              tensor_dim0/1, tile_dim0/1, tensor_dim0_stride
// ---------------------------------------------------------------------------
__device__ __forceinline__ void tdm_load_2d(unsigned lds_off, const void* gptr,
                                            unsigned tensor_d0, unsigned tensor_d1,
                                            unsigned tile_d0, unsigned tile_d1,
                                            unsigned long long stride0) {
    const unsigned long long ga = (unsigned long long)(uintptr_t)gptr;
    u32x4 g0;
    g0[0] = 1u;                                        // count=1 (valid user D#)
    g0[1] = lds_off;                                   // LDS byte address
    g0[2] = (unsigned)(ga & 0xffffffffu);              // global_addr[31:0]
    g0[3] = (unsigned)((ga >> 32) & 0x01ffffffu)       // global_addr[56:32]
          | (2u << 30);                                // type = 2 ("image")
    i32x8 g1;
    g1[0] = (int)((1u << 16)                           // data_size = 1 -> 2 bytes
                | (1u << 20)                           // pad_enable
                | (3u << 22)                           // pad_interval: 16 DWORDs
                | (3u << 25));                         // pad_amount: 4 DWORDs (16B)
    g1[1] = (int)((tensor_d0 & 0xffffu) << 16);        // tensor_dim0[15:0] (atomic addr 0)
    g1[2] = (int)((tensor_d0 >> 16) | ((tensor_d1 & 0xffffu) << 16));
    g1[3] = (int)((tensor_d1 >> 16) | (tile_d0 << 16));// tile_dim0
    g1[4] = (int)tile_d1;                              // tile_dim1 (tile_dim2 = 0)
    g1[5] = (int)(stride0 & 0xffffffffu);              // tensor_dim0_stride[31:0]
    g1[6] = (int)((stride0 >> 32) & 0xffffu);          // stride0[47:32], stride1=0
    g1[7] = 0;
    const i32x4 z4 = {0, 0, 0, 0};                     // D# groups 2/3 (unused, 2-D)
    const i32x8 z8 = {0, 0, 0, 0, 0, 0, 0, 0};         // extra group (clang-23 form)
    __builtin_amdgcn_tensor_load_to_lds(g0, g1, z4, z4, z8, 0);
}

// ---------------------------------------------------------------------------
// Weight convert: fp32 [K,N] row-major  ->  bf16 [N,K] row-major (transposed)
// ---------------------------------------------------------------------------
__global__ __launch_bounds__(256)
void transpose_f32_to_bf16(const float* __restrict__ in, __bf16* __restrict__ out,
                           int K, int N) {
    __shared__ float t[32][33];
    const int n0 = blockIdx.x << 5, k0 = blockIdx.y << 5;
    const int tx = threadIdx.x & 31, ty = threadIdx.x >> 5;   // 32 x 8
#pragma unroll
    for (int j = 0; j < 4; ++j)
        t[ty + j * 8][tx] = in[(size_t)(k0 + ty + j * 8) * N + n0 + tx];
    __syncthreads();
#pragma unroll
    for (int j = 0; j < 4; ++j)
        out[(size_t)(n0 + ty + j * 8) * K + k0 + tx] = (__bf16)t[tx][ty + j * 8];
}

// ---------------------------------------------------------------------------
// LayerNorm (fp32 in) -> bf16 out.  One row (D=1024) per block of 256 threads.
// ---------------------------------------------------------------------------
__global__ __launch_bounds__(256)
void layernorm_to_bf16(const float* __restrict__ x, const float* __restrict__ g,
                       const float* __restrict__ bta, __bf16* __restrict__ out, int D) {
    __shared__ float red[256];
    __shared__ float red2[256];
    const size_t base = (size_t)blockIdx.x * D;
    float s = 0.f, s2 = 0.f;
    for (int i = threadIdx.x; i < D; i += 256) {
        const float v = x[base + i];
        s += v; s2 += v * v;
    }
    red[threadIdx.x] = s; red2[threadIdx.x] = s2;
    __syncthreads();
    for (int off = 128; off > 0; off >>= 1) {
        if (threadIdx.x < off) {
            red[threadIdx.x]  += red[threadIdx.x + off];
            red2[threadIdx.x] += red2[threadIdx.x + off];
        }
        __syncthreads();
    }
    const float mean = red[0] / (float)D;
    const float var  = red2[0] / (float)D - mean * mean;
    const float rstd = rsqrtf(var + 1e-5f);
    for (int i = threadIdx.x; i < D; i += 256) {
        const float v = (x[base + i] - mean) * rstd * g[i] + bta[i];
        out[base + i] = (__bf16)v;
    }
}

// ---------------------------------------------------------------------------
// WMMA GEMM: C[M,N] = A[M,K](bf16,row) x BT[N,K](bf16,row) + bias.
// A/B tiles staged by the Tensor Data Mover (wave 0 issues, TENSORcnt tracked),
// double-buffered in LDS; compute waves read b128 fragments from padded LDS.
// Epilogues:
//   EPI 0: bf16 row-major out (optional exact GELU)      -> FFN1
//   EPI 1: bf16 scatter to [B,H,S,HD]                    -> Q,K
//   EPI 2: bf16 scatter to [B,H,HD,S] (V transposed)     -> V
//   EPI 3: fp32 out + fp32 residual                      -> O-proj, FFN2
// Block: 256 thr (8 wave32), tile 128x64, BK=32. Wave w -> 16-row strip.
// ---------------------------------------------------------------------------
template <int EPI, bool GELU>
__global__ __launch_bounds__(256)
void gemm_wmma(const __bf16* __restrict__ A, const __bf16* __restrict__ BT,
               const float* __restrict__ bias, const float* __restrict__ resid,
               void* __restrict__ outp, int M, int N, int K, int Hn, int Sn) {
    __shared__ __bf16 sA[2][128 * LDA];
    __shared__ __bf16 sB[2][64 * LDA];
    const int tid = threadIdx.x;
    const int wave = tid >> 5, lane = tid & 31;
    const int lhalf = lane & 15, hi = lane >> 4;
    const int row0 = blockIdx.y << 7, col0 = blockIdx.x << 6;

    v8f acc[4] = {};

    const int kTiles = K >> 5;

    // TDM issue helper: tile kt -> buffer kt&1 (two tensor loads: A then B)
    auto issue = [&](int kt) {
        const int k0 = kt << 5;
        const int b = kt & 1;
        tdm_load_2d((unsigned)(uintptr_t)&sA[b][0],
                    A + (size_t)row0 * K + k0,
                    (unsigned)K, 128u, 32u, 128u, (unsigned long long)K);
        tdm_load_2d((unsigned)(uintptr_t)&sB[b][0],
                    BT + (size_t)col0 * K + k0,
                    (unsigned)K, 64u, 32u, 64u, (unsigned long long)K);
    };

    if (wave == 0) issue(0);

    for (int kt = 0; kt < kTiles; ++kt) {
        if (wave == 0) {
            if (kt + 1 < kTiles) {
                issue(kt + 1);                               // prefetch next tiles
                __builtin_amdgcn_s_wait_tensorcnt(2);        // wait for tile kt pair
            } else {
                __builtin_amdgcn_s_wait_tensorcnt(0);
            }
        }
        __syncthreads();   // buffer kt&1 ready for everyone

        const __bf16* bufA = &sA[kt & 1][0];
        const __bf16* bufB = &sB[kt & 1][0];

        // A fragment (16x32): lane = row, 16 consecutive K at hi*16
        v16bf a;
        {
            const int ar = (wave << 4) + lhalf, kb = hi << 4;
            ((v8bf*)&a)[0] = *(const v8bf*)&bufA[ar * LDA + kb];
            ((v8bf*)&a)[1] = *(const v8bf*)&bufA[ar * LDA + kb + 8];
        }
#pragma unroll
        for (int nt = 0; nt < 4; ++nt) {
            // B fragment (32x16): lane = column, 16 consecutive K at hi*16
            v16bf b;
            const int br = (nt << 4) + lhalf, kb2 = hi << 4;
            ((v8bf*)&b)[0] = *(const v8bf*)&bufB[br * LDA + kb2];
            ((v8bf*)&b)[1] = *(const v8bf*)&bufB[br * LDA + kb2 + 8];
            acc[nt] = __builtin_amdgcn_wmma_f32_16x16x32_bf16(
                false, a, false, b, (short)0, acc[nt], false, false);
        }
        __syncthreads();   // done reading buffer kt&1; TDM may overwrite next iter
    }

    // Epilogue. C layout: lane -> col = lhalf, vgpr r -> row = r + 8*hi.
    const int wrow = wave << 4;
#pragma unroll
    for (int nt = 0; nt < 4; ++nt) {
#pragma unroll
        for (int r = 0; r < 8; ++r) {
            const int row = row0 + wrow + r + (hi << 3);
            const int col = col0 + (nt << 4) + lhalf;
            float v = acc[nt][r] + bias[col];
            if (EPI == 0) {
                if (GELU) v = gelu_exact(v);
                ((__bf16*)outp)[(size_t)row * N + col] = (__bf16)v;
            } else if (EPI == 1) {
                const int bb = row / Sn, s = row - bb * Sn;
                const int h = col >> 6, hd = col & 63;
                ((__bf16*)outp)[(((size_t)bb * Hn + h) * Sn + s) * 64 + hd] = (__bf16)v;
            } else if (EPI == 2) {
                const int bb = row / Sn, s = row - bb * Sn;
                const int h = col >> 6, hd = col & 63;
                ((__bf16*)outp)[(((size_t)bb * Hn + h) * 64 + hd) * Sn + s] = (__bf16)v;
            } else {
                ((float*)outp)[(size_t)row * N + col] =
                    v + resid[(size_t)row * N + col];
            }
        }
    }
}

// ---------------------------------------------------------------------------
// Flash-style MHA. q,k: [B,H,S,64] bf16; vt: [B,H,64,S] bf16 (V transposed).
// Block = 128 thr (4 waves); each wave owns a 16-row Q strip, streams keys in
// 32-wide tiles with online softmax; P re-laid out C->A via per-wave LDS.
// out: [B,S,D] bf16.
// ---------------------------------------------------------------------------
__global__ __launch_bounds__(128)
void attn_wmma(const __bf16* __restrict__ q, const __bf16* __restrict__ k,
               const __bf16* __restrict__ vt, __bf16* __restrict__ out,
               int Hn, int Sn, int Dn) {
    __shared__ __bf16 sP[4][16 * LDA];
    const int tid = threadIdx.x, wave = tid >> 5, lane = tid & 31;
    const int lhalf = lane & 15, hi = lane >> 4;
    const int bh = blockIdx.y;
    const int bb = bh / Hn, h = bh - bb * Hn;
    const int qrow0 = (blockIdx.x << 6) + (wave << 4);
    const __bf16* qb  = q  + (size_t)bh * Sn * 64;
    const __bf16* kbp = k  + (size_t)bh * Sn * 64;
    const __bf16* vb  = vt + (size_t)bh * 64 * Sn;

    // Q fragments for this 16-row strip: a0 = hd 0..31, a1 = hd 32..63
    v16bf a0, a1;
    {
        const int ar = qrow0 + lhalf, kb = hi << 4;
        ((v8bf*)&a0)[0] = *(const v8bf*)&qb[(size_t)ar * 64 + kb];
        ((v8bf*)&a0)[1] = *(const v8bf*)&qb[(size_t)ar * 64 + kb + 8];
        ((v8bf*)&a1)[0] = *(const v8bf*)&qb[(size_t)ar * 64 + 32 + kb];
        ((v8bf*)&a1)[1] = *(const v8bf*)&qb[(size_t)ar * 64 + 32 + kb + 8];
    }

    v8f o[4] = {};
    float m[8], l[8];
#pragma unroll
    for (int r = 0; r < 8; ++r) { m[r] = -1e30f; l[r] = 0.f; }
    const float scale = 0.125f;   // 1/sqrt(64)

    for (int kt = 0; kt < (Sn >> 5); ++kt) {
        // ---- scores for a 16x32 tile (two 16x16 WMMA-N tiles) ----
        v8f s0 = {}, s1 = {};
#pragma unroll
        for (int sub = 0; sub < 2; ++sub) {
            const int key = (kt << 5) + (sub << 4) + lhalf;  // lane = key column
            const int kbase = hi << 4;
            v16bf b0, b1;   // K^T fragments: contiguous hd per key row
            ((v8bf*)&b0)[0] = *(const v8bf*)&kbp[(size_t)key * 64 + kbase];
            ((v8bf*)&b0)[1] = *(const v8bf*)&kbp[(size_t)key * 64 + kbase + 8];
            ((v8bf*)&b1)[0] = *(const v8bf*)&kbp[(size_t)key * 64 + 32 + kbase];
            ((v8bf*)&b1)[1] = *(const v8bf*)&kbp[(size_t)key * 64 + 32 + kbase + 8];
            v8f s = {};
            s = __builtin_amdgcn_wmma_f32_16x16x32_bf16(false, a0, false, b0, (short)0, s, false, false);
            s = __builtin_amdgcn_wmma_f32_16x16x32_bf16(false, a1, false, b1, (short)0, s, false, false);
            if (sub == 0) s0 = s; else s1 = s;
        }

        // ---- online softmax over the 32 keys (rows live in 16-lane halves) ----
        float p0[8], p1[8];
#pragma unroll
        for (int r = 0; r < 8; ++r) {
            const float x0 = s0[r] * scale, x1 = s1[r] * scale;
            float tmax = fmaxf(x0, x1);
#pragma unroll
            for (int off = 1; off <= 8; off <<= 1)
                tmax = fmaxf(tmax, __shfl_xor(tmax, off, 32));
            const float nm  = fmaxf(m[r], tmax);
            const float fac = __expf(m[r] - nm);
            const float e0 = __expf(x0 - nm), e1 = __expf(x1 - nm);
            float rs = e0 + e1;
#pragma unroll
            for (int off = 1; off <= 8; off <<= 1)
                rs += __shfl_xor(rs, off, 32);
            l[r] = l[r] * fac + rs;
            m[r] = nm;
            p0[r] = e0; p1[r] = e1;
#pragma unroll
            for (int nt = 0; nt < 4; ++nt) o[nt][r] *= fac;
        }

        // ---- P: C layout -> LDS (bf16 16x32) -> A fragment ----
#pragma unroll
        for (int r = 0; r < 8; ++r) {
            const int prow = r + (hi << 3);
            sP[wave][prow * LDA + lhalf]      = (__bf16)p0[r];
            sP[wave][prow * LDA + 16 + lhalf] = (__bf16)p1[r];
        }
        v16bf pa;
        {
            const int kb = hi << 4;
            ((v8bf*)&pa)[0] = *(v8bf*)&sP[wave][lhalf * LDA + kb];
            ((v8bf*)&pa)[1] = *(v8bf*)&sP[wave][lhalf * LDA + kb + 8];
        }

        // ---- O += P x V   (Vt gives contiguous keys per hd row) ----
#pragma unroll
        for (int nt = 0; nt < 4; ++nt) {
            v16bf bv;
            const int hd = (nt << 4) + lhalf;
            const size_t kk = (size_t)(kt << 5) + (hi << 4);
            ((v8bf*)&bv)[0] = *(const v8bf*)&vb[(size_t)hd * Sn + kk];
            ((v8bf*)&bv)[1] = *(const v8bf*)&vb[(size_t)hd * Sn + kk + 8];
            o[nt] = __builtin_amdgcn_wmma_f32_16x16x32_bf16(
                false, pa, false, bv, (short)0, o[nt], false, false);
        }
    }

    // ---- normalize and scatter to [B,S,D] bf16 ----
#pragma unroll
    for (int nt = 0; nt < 4; ++nt) {
#pragma unroll
        for (int r = 0; r < 8; ++r) {
            const int srow = qrow0 + r + (hi << 3);
            const float ov = o[nt][r] / l[r];
            out[((size_t)bb * Sn + srow) * Dn + (h << 6) + (nt << 4) + lhalf] = (__bf16)ov;
        }
    }
}

// ---------------------------------------------------------------------------
// Host orchestration
// ---------------------------------------------------------------------------
extern "C" void kernel_launch(void* const* d_in, const int* in_sizes, int n_in,
                              void* d_out, int out_size, void* d_ws, size_t ws_size,
                              hipStream_t stream) {
    (void)in_sizes; (void)n_in; (void)out_size; (void)ws_size;
    const int Bc = 4, Sc = 2048, Dc = 1024, Hc = 16, DFc = 4096;
    const int Mr = Bc * Sc;  // 8192 token rows

    const float* x   = (const float*)d_in[0];
    const float* Wq  = (const float*)d_in[1];
    const float* bq  = (const float*)d_in[2];
    const float* Wk  = (const float*)d_in[3];
    const float* bk  = (const float*)d_in[4];
    const float* Wv  = (const float*)d_in[5];
    const float* bv  = (const float*)d_in[6];
    const float* Wo  = (const float*)d_in[7];
    const float* bo  = (const float*)d_in[8];
    const float* g1  = (const float*)d_in[9];
    const float* be1 = (const float*)d_in[10];
    const float* g2  = (const float*)d_in[11];
    const float* be2 = (const float*)d_in[12];
    const float* W1  = (const float*)d_in[13];
    const float* b1  = (const float*)d_in[14];
    const float* W2  = (const float*)d_in[15];
    const float* b2  = (const float*)d_in[16];

    char* ws = (char*)d_ws;
    size_t off = 0;
    auto alloc = [&](size_t bytes) -> void* {
        void* p = ws + off;
        off = (off + bytes + 255) & ~(size_t)255;
        return p;
    };

    __bf16* wqT  = (__bf16*)alloc((size_t)Dc * Dc * 2);
    __bf16* wkT  = (__bf16*)alloc((size_t)Dc * Dc * 2);
    __bf16* wvT  = (__bf16*)alloc((size_t)Dc * Dc * 2);
    __bf16* woT  = (__bf16*)alloc((size_t)Dc * Dc * 2);
    __bf16* w1T  = (__bf16*)alloc((size_t)Dc * DFc * 2);
    __bf16* w2T  = (__bf16*)alloc((size_t)DFc * Dc * 2);
    __bf16* xn   = (__bf16*)alloc((size_t)Mr * Dc * 2);
    __bf16* qb   = (__bf16*)alloc((size_t)Mr * Dc * 2);
    __bf16* kb   = (__bf16*)alloc((size_t)Mr * Dc * 2);
    __bf16* vtb  = (__bf16*)alloc((size_t)Mr * Dc * 2);
    __bf16* attn = (__bf16*)alloc((size_t)Mr * Dc * 2);
    float*  x1   = (float*) alloc((size_t)Mr * Dc * 4);
    __bf16* xn2  = (__bf16*)alloc((size_t)Mr * Dc * 2);
    __bf16* hb   = (__bf16*)alloc((size_t)Mr * DFc * 2);

    // 1) weights -> bf16 transposed [N,K]
    transpose_f32_to_bf16<<<dim3(Dc / 32, Dc / 32),  256, 0, stream>>>(Wq, wqT, Dc, Dc);
    transpose_f32_to_bf16<<<dim3(Dc / 32, Dc / 32),  256, 0, stream>>>(Wk, wkT, Dc, Dc);
    transpose_f32_to_bf16<<<dim3(Dc / 32, Dc / 32),  256, 0, stream>>>(Wv, wvT, Dc, Dc);
    transpose_f32_to_bf16<<<dim3(Dc / 32, Dc / 32),  256, 0, stream>>>(Wo, woT, Dc, Dc);
    transpose_f32_to_bf16<<<dim3(DFc / 32, Dc / 32), 256, 0, stream>>>(W1, w1T, Dc, DFc);
    transpose_f32_to_bf16<<<dim3(Dc / 32, DFc / 32), 256, 0, stream>>>(W2, w2T, DFc, Dc);

    // 2) LN1 -> bf16
    layernorm_to_bf16<<<Mr, 256, 0, stream>>>(x, g1, be1, xn, Dc);

    // 3) Q, K (scatter [B,H,S,HD]) and V (scatter [B,H,HD,S])
    const dim3 gD(Dc / 64, Mr / 128);
    gemm_wmma<1, false><<<gD, 256, 0, stream>>>(xn, wqT, bq, nullptr, qb,  Mr, Dc, Dc, Hc, Sc);
    gemm_wmma<1, false><<<gD, 256, 0, stream>>>(xn, wkT, bk, nullptr, kb,  Mr, Dc, Dc, Hc, Sc);
    gemm_wmma<2, false><<<gD, 256, 0, stream>>>(xn, wvT, bv, nullptr, vtb, Mr, Dc, Dc, Hc, Sc);

    // 4) attention -> [B,S,D] bf16
    attn_wmma<<<dim3(Sc / 64, Bc * Hc), 128, 0, stream>>>(qb, kb, vtb, attn, Hc, Sc, Dc);

    // 5) O-proj + residual(x) -> x1 fp32
    gemm_wmma<3, false><<<gD, 256, 0, stream>>>(attn, woT, bo, x, x1, Mr, Dc, Dc, Hc, Sc);

    // 6) LN2 -> bf16
    layernorm_to_bf16<<<Mr, 256, 0, stream>>>(x1, g2, be2, xn2, Dc);

    // 7) FFN1 + exact GELU -> bf16 [Mr, 4096]
    gemm_wmma<0, true><<<dim3(DFc / 64, Mr / 128), 256, 0, stream>>>(
        xn2, w1T, b1, nullptr, hb, Mr, DFc, Dc, Hc, Sc);

    // 8) FFN2 + residual(x1) -> d_out fp32
    gemm_wmma<3, false><<<gD, 256, 0, stream>>>(hb, w2T, b2, x1, d_out, Mr, Dc, DFc, Hc, Sc);
}
